// BiUNet_Binaryconect_body_58025008169033
// MI455X (gfx1250) — matrix operand order repaired and checked by reading, not compiled, not run on hardware
//
#include <hip/hip_runtime.h>
#include <math.h>

typedef _Float16 f16;
typedef __attribute__((ext_vector_type(8)))  _Float16 half8;
typedef __attribute__((ext_vector_type(16))) _Float16 half16;
typedef __attribute__((ext_vector_type(8)))  float    v8f;

#define EPSF 1e-5f

static __device__ inline half16 cat8(half8 lo, half8 hi) {
  return __builtin_shufflevector(lo, hi, 0,1,2,3,4,5,6,7,8,9,10,11,12,13,14,15);
}
static __device__ inline v8f wmma16(half16 a, half16 b, v8f c) {
  // D = A(16x32 f16) * B(32x16 f16) + C(f32)
  return __builtin_amdgcn_wmma_f32_16x16x32_f16(false, a, false, b, (short)0, c, false, false);
}

// Async DMA: global -> LDS, 16 bytes per lane, tracked by ASYNCcnt.
static __device__ inline void async_ld_b128(unsigned lds_addr, const void* gaddr) {
  asm volatile("global_load_async_to_lds_b128 %0, %1, off"
               :: "v"(lds_addr), "v"((unsigned long long)(size_t)gaddr) : "memory");
}
static __device__ inline void wait_async3() {
  asm volatile("s_wait_asynccnt 0x3" ::: "memory");
}
static __device__ inline void wait_async0() {
  asm volatile("s_wait_asynccnt 0x0" ::: "memory");
}

// ---------------------------------------------------------------------------
// Generic conv-as-GEMM via WMMA.  Out[p, n] = sum_{tap,k} X[shift(p,tap), k] * W[tap][n][k]
// X: f16 [P][Kp] (NHWC, padded channels, zero pad). W: f16 [taps][Np][Kp] (+-1, zero pad).
// Block: 256 threads = 8 waves; block tile = 128 pixels x 64 out-channels.
// Double-buffered LDS staged with global_load_async_to_lds_b128:
//   per stage each thread issues exactly 3 async loads (2 x A, 1 x B), so each
//   wave's ASYNCcnt rises by 3 per stage; s_wait_asynccnt 3 after issuing the
//   next stage guarantees (in-order completion) the current stage is resident.
// ---------------------------------------------------------------------------
__global__ __launch_bounds__(256)
void k_conv_wmma(const f16* __restrict__ X, const f16* __restrict__ Wt,
                 float* __restrict__ Y, int P, int H, int W,
                 int Kp, int N, int Np, int taps)
{
  __shared__ f16 Al[2][128 * 32];   // 16 KB
  __shared__ f16 Bl[2][64 * 32];    //  8 KB
  const int tid  = threadIdx.x;
  const int wave = tid >> 5;
  const int lane = tid & 31;
  const int p0 = blockIdx.x * 128;
  const int n0 = blockIdx.y * 64;
  const int PB = H * W;

  // A staging: thread -> row ar, 16-half segment aseg (two b128 chunks).
  const int ar   = tid >> 1;
  const int aseg = tid & 1;
  const int p  = p0 + ar;
  const int b  = p / PB;
  const int rm = p - b * PB;
  const int hh = rm / W;
  const int ww = rm - hh * W;
  // B staging: thread -> row bn, 8-half chunk bch (one b128 chunk).
  const int bnr = tid >> 2;
  const int bch = tid & 3;

  const int KT = Kp >> 5;        // K-steps per tap
  const int S  = taps * KT;      // total stages

  v8f acc0 = {0.f,0.f,0.f,0.f,0.f,0.f,0.f,0.f};
  v8f acc1 = acc0, acc2 = acc0, acc3 = acc0;

  auto issue = [&](int s, int buf) {
    const int tap = s / KT;
    const int kt  = (s - tap * KT) << 5;
    const int dy = (taps == 9) ? (tap / 3 - 1) : 0;
    const int dx = (taps == 9) ? (tap % 3 - 1) : 0;
    // clamp (garbage rows zero-patched at consume time) to keep the per-wave
    // async-instruction count uniform.
    const int hn = min(max(hh + dy, 0), H - 1);
    const int wn = min(max(ww + dx, 0), W - 1);
    const f16* ga = X + ((size_t)(b * PB + hn * W + wn)) * (size_t)Kp + kt + aseg * 16;
    const unsigned la = (unsigned)(size_t)&Al[buf][ar * 32 + aseg * 16];
    async_ld_b128(la,      ga);
    async_ld_b128(la + 16, ga + 8);
    const f16* gb = Wt + ((size_t)tap * Np + (n0 + bnr)) * (size_t)Kp + kt + bch * 8;
    const unsigned lb = (unsigned)(size_t)&Bl[buf][bnr * 32 + bch * 8];
    async_ld_b128(lb, gb);
  };
  auto fixup = [&](int s, int buf) {
    const int tap = s / KT;
    const int dy = (taps == 9) ? (tap / 3 - 1) : 0;
    const int dx = (taps == 9) ? (tap % 3 - 1) : 0;
    const int hn = hh + dy, wn = ww + dx;
    const bool valid = (hn >= 0) && (hn < H) && (wn >= 0) && (wn < W);
    if (!valid) {
      uint4 z = make_uint4(0, 0, 0, 0);
      uint4* d = (uint4*)&Al[buf][ar * 32 + aseg * 16];
      d[0] = z; d[1] = z;
    }
  };

  issue(0, 0);
  for (int s = 0; s < S; ++s) {
    const int buf = s & 1;
    if (s + 1 < S) { issue(s + 1, buf ^ 1); wait_async3(); }
    else           { wait_async0(); }
    fixup(s, buf);
    __syncthreads();

    // A fragment: lane m=lane&15, contiguous K run of 16 at 16*(lane>>4).
    const int m  = lane & 15;
    const int ao = (lane >> 4) * 16;
    const f16* ab = &Al[buf][(wave * 16 + m) * 32];
    half16 af = cat8(*(const half8*)(ab + ao), *(const half8*)(ab + ao + 8));
    // B fragments: lane n=lane&15, contiguous K run of 16 at 16*(lane>>4).
    const int bo = (lane >> 4) * 16;
    const f16* b0 = &Bl[buf][m * 32 + bo];
    const f16* b1 = &Bl[buf][(16 + m) * 32 + bo];
    const f16* b2 = &Bl[buf][(32 + m) * 32 + bo];
    const f16* b3 = &Bl[buf][(48 + m) * 32 + bo];
    half16 bf0 = cat8(*(const half8*)b0, *(const half8*)(b0 + 8));
    half16 bf1 = cat8(*(const half8*)b1, *(const half8*)(b1 + 8));
    half16 bf2 = cat8(*(const half8*)b2, *(const half8*)(b2 + 8));
    half16 bf3 = cat8(*(const half8*)b3, *(const half8*)(b3 + 8));

    acc0 = wmma16(af, bf0, acc0);
    acc1 = wmma16(af, bf1, acc1);
    acc2 = wmma16(af, bf2, acc2);
    acc3 = wmma16(af, bf3, acc3);
    __syncthreads();
  }

  // C/D layout: lane n = lane&15, VGPR r -> row r + 8*(lane>>4).
  const int nn = lane & 15;
  const int mh = (lane >> 4) * 8;
  for (int r = 0; r < 8; ++r) {
    const size_t row = (size_t)(p0 + wave * 16 + mh + r) * (size_t)N;
    if (n0 + nn < N)      Y[row + n0 + nn]      = acc0[r];
    if (n0 + 16 + nn < N) Y[row + n0 + 16 + nn] = acc1[r];
    if (n0 + 32 + nn < N) Y[row + n0 + 32 + nn] = acc2[r];
    if (n0 + 48 + nn < N) Y[row + n0 + 48 + nn] = acc3[r];
  }
}

// ---------------------------------------------------------------------------
// Weight binarization: sign(sign(w)+0.01) == (w>=0 ? +1 : -1), padded f16.
// Layout out: [taps][Np][Kp].  In: [Co][Ci][kh][kw].
// ---------------------------------------------------------------------------
__global__ void k_binw(const float* __restrict__ w, f16* __restrict__ o,
                       int taps, int Co, int Ci, int Np, int Kp)
{
  int idx = blockIdx.x * blockDim.x + threadIdx.x;
  int total = taps * Np * Kp;
  if (idx >= total) return;
  int t = idx / (Np * Kp);
  int r = idx - t * (Np * Kp);
  int n = r / Kp;
  int k = r - n * Kp;
  f16 v = (f16)0;
  if (n < Co && k < Ci) {
    float x = w[((size_t)n * Ci + k) * taps + t];
    v = (x >= 0.f) ? (f16)1.f : (f16)-1.f;
  }
  o[idx] = v;
}

// grouped 3x3 weights: in [m2][4][3][3] -> out [m2][9][4]
__global__ void k_bing(const float* __restrict__ w, f16* __restrict__ o, int m2)
{
  int idx = blockIdx.x * blockDim.x + threadIdx.x;
  if (idx >= m2 * 36) return;
  int co = idx / 36;
  int r  = idx - co * 36;
  int tap = r >> 2, ci = r & 3;
  float x = w[((size_t)co * 4 + ci) * 9 + tap];
  o[idx] = (x >= 0.f) ? (f16)1.f : (f16)-1.f;
}

// ---------------------------------------------------------------------------
// Per-channel sum / sumsq for training-mode BatchNorm.  One block per channel.
// ---------------------------------------------------------------------------
__global__ void k_stats(const float* __restrict__ Y, float* __restrict__ st, int P, int N)
{
  __shared__ float s1[256], s2[256];
  const int c = blockIdx.x;
  float a = 0.f, b = 0.f;
  for (int p = threadIdx.x; p < P; p += 256) {
    float v = Y[(size_t)p * N + c];
    a += v; b += v * v;
  }
  s1[threadIdx.x] = a; s2[threadIdx.x] = b; __syncthreads();
  for (int o = 128; o > 0; o >>= 1) {
    if (threadIdx.x < o) { s1[threadIdx.x] += s1[threadIdx.x + o];
                           s2[threadIdx.x] += s2[threadIdx.x + o]; }
    __syncthreads();
  }
  if (threadIdx.x == 0) { st[c] = s1[0]; st[N + c] = s2[0]; }
}

// BN (batch stats) + ReLU [+ exact GELU] [+ sign], emits f32 and/or padded f16.
__global__ void k_bnact(const float* __restrict__ Y, const float* __restrict__ st,
                        const float* __restrict__ gg, const float* __restrict__ bb,
                        float* __restrict__ of32, f16* __restrict__ of16,
                        int P, int N, int Kp, int flags, float invP)
{
  const int CS = of16 ? Kp : N;
  const long long n = (long long)P * CS;
  long long idx = (long long)blockIdx.x * blockDim.x + threadIdx.x;
  if (idx >= n) return;
  const int c = (int)(idx % CS);
  const long long pp = idx / CS;
  if (c >= N) { of16[pp * Kp + c] = (f16)0; return; }
  const float mu  = st[c] * invP;
  const float var = st[N + c] * invP - mu * mu;
  float v = (Y[pp * N + c] - mu) * rsqrtf(var + EPSF) * gg[c] + bb[c];
  v = fmaxf(v, 0.f);
  if (flags & 1) v = 0.5f * v * (1.f + erff(v * 0.70710678118654752f));
  if (of32) of32[pp * N + c] = v;
  if (of16) of16[pp * Kp + c] =
      (flags & 2) ? (f16)((v > 0.f) ? 1.f : ((v < 0.f) ? -1.f : 0.f)) : (f16)v;
}

// LayerNorm over channels (NHWC), emits padded f16 for next GEMM.
__global__ void k_ln(const float* __restrict__ X, const float* __restrict__ g,
                     const float* __restrict__ b, f16* __restrict__ O,
                     int P, int C, int Kp)
{
  int p = blockIdx.x * blockDim.x + threadIdx.x;
  if (p >= P) return;
  const float* x = X + (size_t)p * C;
  float s = 0.f;
  for (int c = 0; c < C; ++c) s += x[c];
  float mu = s / C;
  float q = 0.f;
  for (int c = 0; c < C; ++c) { float d = x[c] - mu; q += d * d; }
  float rs = rsqrtf(q / C + EPSF);
  f16* o = O + (size_t)p * Kp;
  for (int c = 0; c < C; ++c) o[c] = (f16)((x[c] - mu) * rs * g[c] + b[c]);
  for (int c = C; c < Kp; ++c) o[c] = (f16)0;
}

// Grouped 3x3 conv, group size 4 (c3): +-1 activations x +-1 weights.
__global__ void k_gconv3(const f16* __restrict__ S, const f16* __restrict__ W3,
                         float* __restrict__ Y, int P, int H, int W, int m2, int Kp)
{
  long long idx = (long long)blockIdx.x * blockDim.x + threadIdx.x;
  if (idx >= (long long)P * m2) return;
  const int co = (int)(idx % m2);
  const int p  = (int)(idx / m2);
  const int g4 = co & ~3;
  const int PB = H * W;
  const int b  = p / PB;
  const int rm = p - b * PB;
  const int h  = rm / W;
  const int w  = rm - h * W;
  float acc = 0.f;
  for (int ky = 0; ky < 3; ++ky) {
    int hn = h + ky - 1;
    if (hn < 0 || hn >= H) continue;
    for (int kx = 0; kx < 3; ++kx) {
      int wn = w + kx - 1;
      if (wn < 0 || wn >= W) continue;
      const int tap = ky * 3 + kx;
      const size_t pn = (size_t)(b * PB + hn * W + wn) * (size_t)Kp + g4;
      const size_t wb = (size_t)co * 36 + tap * 4;
      for (int ci = 0; ci < 4; ++ci)
        acc += (float)S[pn + ci] * (float)W3[wb + ci];
    }
  }
  Y[idx] = acc;
}

// sign(x) from NCHW f32 -> padded f16 NHWC (bact for embedding input)
__global__ void k_sign_nchw(const float* __restrict__ x, f16* __restrict__ O,
                            int P, int C, int Kp, int HW)
{
  long long idx = (long long)blockIdx.x * blockDim.x + threadIdx.x;
  if (idx >= (long long)P * Kp) return;
  int c = (int)(idx % Kp);
  int p = (int)(idx / Kp);
  if (c >= C) { O[idx] = (f16)0; return; }
  int b = p / HW, hw = p - b * HW;
  float v = x[((size_t)b * C + c) * HW + hw];
  O[idx] = (f16)((v > 0.f) ? 1.f : ((v < 0.f) ? -1.f : 0.f));
}

// sign(x) from NHWC f32 -> padded f16 (bact for mapping input)
__global__ void k_sign_f32(const float* __restrict__ X, f16* __restrict__ O,
                           int P, int C, int Kp)
{
  long long idx = (long long)blockIdx.x * blockDim.x + threadIdx.x;
  if (idx >= (long long)P * Kp) return;
  int c = (int)(idx % Kp);
  int p = (int)(idx / Kp);
  if (c >= C) { O[idx] = (f16)0; return; }
  float v = X[(size_t)p * C + c];
  O[idx] = (f16)((v > 0.f) ? 1.f : ((v < 0.f) ? -1.f : 0.f));
}

// 2x2 avgpool NHWC f32 -> padded f16
__global__ void k_avgpool(const float* __restrict__ X, f16* __restrict__ O,
                          int Pout, int C, int Kp, int Ho, int Wo)
{
  long long idx = (long long)blockIdx.x * blockDim.x + threadIdx.x;
  if (idx >= (long long)Pout * Kp) return;
  int c  = (int)(idx % Kp);
  int po = (int)(idx / Kp);
  if (c >= C) { O[idx] = (f16)0; return; }
  int b  = po / (Ho * Wo);
  int rm = po - b * (Ho * Wo);
  int ho = rm / Wo, wo = rm - ho * Wo;
  int Wi = 2 * Wo;
  size_t base = (size_t)((b * 2 * Ho + 2 * ho) * Wi + 2 * wo);
  float v = 0.25f * (X[base * C + c] + X[(base + 1) * C + c] +
                     X[(base + Wi) * C + c] + X[(base + Wi + 1) * C + c]);
  O[idx] = (f16)v;
}

// bilinear x2 upsample (align_corners=False) NHWC f32 -> padded f16
__global__ void k_up2(const float* __restrict__ X, f16* __restrict__ O,
                      int Pout, int C, int Kp, int Ho, int Wo)
{
  long long idx = (long long)blockIdx.x * blockDim.x + threadIdx.x;
  if (idx >= (long long)Pout * Kp) return;
  int c  = (int)(idx % Kp);
  int po = (int)(idx / Kp);
  if (c >= C) { O[idx] = (f16)0; return; }
  int b  = po / (Ho * Wo);
  int rm = po - b * (Ho * Wo);
  int ho = rm / Wo, wo = rm - ho * Wo;
  int Hi = Ho >> 1, Wi = Wo >> 1;
  float sh = fminf(fmaxf((ho + 0.5f) * 0.5f - 0.5f, 0.f), (float)(Hi - 1));
  float sw = fminf(fmaxf((wo + 0.5f) * 0.5f - 0.5f, 0.f), (float)(Wi - 1));
  int i0 = (int)sh, j0 = (int)sw;
  int i1 = min(i0 + 1, Hi - 1), j1 = min(j0 + 1, Wi - 1);
  float th = sh - i0, tw = sw - j0;
  size_t rb = (size_t)(b * Hi);
  float v00 = X[((rb + i0) * Wi + j0) * C + c];
  float v01 = X[((rb + i0) * Wi + j1) * C + c];
  float v10 = X[((rb + i1) * Wi + j0) * C + c];
  float v11 = X[((rb + i1) * Wi + j1) * C + c];
  float v = (v00 * (1.f - tw) + v01 * tw) * (1.f - th) +
            (v10 * (1.f - tw) + v11 * tw) * th;
  O[idx] = (f16)v;
}

// channel concat of two NHWC f32 -> padded f16
__global__ void k_concat(const float* __restrict__ A, const float* __restrict__ B,
                         f16* __restrict__ O, int P, int C1, int C2, int Kp)
{
  long long idx = (long long)blockIdx.x * blockDim.x + threadIdx.x;
  if (idx >= (long long)P * Kp) return;
  int c = (int)(idx % Kp);
  int p = (int)(idx / Kp);
  f16 v = (f16)0;
  if (c < C1) v = (f16)A[(size_t)p * C1 + c];
  else if (c < C1 + C2) v = (f16)B[(size_t)p * C2 + (c - C1)];
  O[idx] = v;
}

__global__ void k_add(float* __restrict__ X, const float* __restrict__ Y, long long n)
{
  long long i = (long long)blockIdx.x * blockDim.x + threadIdx.x;
  if (i < n) X[i] += Y[i];
}

// out NCHW = conv_out NHWC + x_in NCHW
__global__ void k_final(const float* __restrict__ Y, const float* __restrict__ xin,
                        float* __restrict__ out, int P, int C, int HW)
{
  long long idx = (long long)blockIdx.x * blockDim.x + threadIdx.x;
  if (idx >= (long long)P * C) return;
  long long chw = (long long)C * HW;
  int b  = (int)(idx / chw);
  long long r = idx - (long long)b * chw;
  int c  = (int)(r / HW);
  int hw = (int)(r - (long long)c * HW);
  int p  = b * HW + hw;
  out[idx] = Y[(size_t)p * C + c] + xin[idx];
}

// ---------------------------------------------------------------------------
// Host orchestration
// ---------------------------------------------------------------------------
extern "C" void kernel_launch(void* const* d_in, const int* in_sizes, int n_in,
                              void* d_out, int out_size, void* d_ws, size_t ws_size,
                              hipStream_t stream)
{
  (void)in_sizes; (void)n_in; (void)out_size; (void)ws_size;
  int cur = 0;
  const float* x_in = (const float*)d_in[cur++];

  struct BNP { const float* g; const float* b; };
  struct CP  { const float* w; BNP bn; };
  struct FFP { CP c1, c2, c3, c4, c5; };
  struct BLK { const float* lg; const float* lb; FFP ff; };

  auto rd   = [&]() { return (const float*)d_in[cur++]; };
  auto rdBN = [&]() { BNP r; r.g = rd(); r.b = rd(); return r; };
  auto rdC  = [&]() { CP r; r.w = rd(); r.bn = rdBN(); return r; };
  auto rdFF = [&]() { FFP f; f.c1 = rdC(); f.c2 = rdC(); f.c3 = rdC();
                      f.c4 = rdC(); f.c5 = rdC(); return f; };
  auto rdBlk= [&]() { BLK b; b.lg = rd(); b.lb = rd(); b.ff = rdFF(); return b; };

  // insertion-order traversal of params pytree
  const float* emb_w = rd();
  BLK e0b[2]; for (int i = 0; i < 2; ++i) e0b[i] = rdBlk();
  CP  down0 = rdC();
  BLK e1b[4]; for (int i = 0; i < 4; ++i) e1b[i] = rdBlk();
  CP  down1 = rdC();
  BLK bo[4];  for (int i = 0; i < 4; ++i) bo[i] = rdBlk();
  CP  up0 = rdC(); CP fus0 = rdC();
  BLK d0b[4]; for (int i = 0; i < 4; ++i) d0b[i] = rdBlk();
  CP  up1 = rdC(); CP fus1 = rdC();
  BLK d1b[2]; for (int i = 0; i < 2; ++i) d1b[i] = rdBlk();
  const float* map_w = rd();

  const int H0 = 256, W0 = 256, P0 = 2 * H0 * W0;   // 131072
  const int H1 = 128, W1 = 128, P1 = 2 * H1 * W1;   // 32768
  const int H2 = 64,  W2 = 64,  P2 = 2 * H2 * W2;   // 8192

  char* wsb = (char*)d_ws; size_t off = 0;
  auto alloc = [&](size_t bytes) -> void* {
    void* p = wsb + off; off = (off + bytes + 255) & ~(size_t)255; return p;
  };

  float* Y   = (float*)alloc((size_t)P0 * 112 * sizeof(float)); // GEMM out temp
  f16*   T16 = (f16*)  alloc((size_t)P0 * 128 * sizeof(f16));   // f16 staging
  float* U   = (float*)alloc((size_t)P0 * 28  * sizeof(float)); // up-branch temp
  float* X0  = (float*)alloc((size_t)P0 * 28  * sizeof(float)); // stream L0
  float* X1  = (float*)alloc((size_t)P1 * 56  * sizeof(float)); // stream L1
  float* X2  = (float*)alloc((size_t)P2 * 112 * sizeof(float)); // stream L2
  float* ST  = (float*)alloc(4096);                             // BN stats

  auto pad32 = [](int v) { return (v + 31) & ~31; };
  auto pad64 = [](int v) { return (v + 63) & ~63; };

  struct CW { f16* w; int taps, N, Np, Kp; };
  auto mk = [&](const float* w, int Co, int Ci, int k) -> CW {
    CW c; c.taps = k * k; c.N = Co; c.Np = pad64(Co); c.Kp = pad32(Ci);
    size_t n = (size_t)c.taps * c.Np * c.Kp;
    c.w = (f16*)alloc(n * sizeof(f16));
    k_binw<<<(unsigned)((n + 255) / 256), 256, 0, stream>>>(w, c.w, c.taps, Co, Ci, c.Np, c.Kp);
    return c;
  };
  auto mkg = [&](const float* w, int m2) -> f16* {
    size_t n = (size_t)m2 * 36;
    f16* o = (f16*)alloc(n * sizeof(f16));
    k_bing<<<(unsigned)((n + 255) / 256), 256, 0, stream>>>(w, o, m2);
    return o;
  };
  auto conv = [&](const f16* Xf, const CW& c, float* Yo, int P, int H, int W) {
    dim3 g(P / 128, c.Np / 64);
    k_conv_wmma<<<g, 256, 0, stream>>>(Xf, c.w, Yo, P, H, W, c.Kp, c.N, c.Np, c.taps);
  };
  auto bnact = [&](float* Yi, BNP bn, int P, int N, int Kp,
                   float* o32, f16* o16, int flags) {
    k_stats<<<N, 256, 0, stream>>>(Yi, ST, P, N);
    long long n = (long long)P * (o16 ? Kp : N);
    k_bnact<<<(unsigned)((n + 255) / 256), 256, 0, stream>>>(
        Yi, ST, bn.g, bn.b, o32, o16, P, N, Kp, flags, 1.f / (float)P);
  };

  auto run_blk = [&](float* Xs, int P, int H, int W, int d, const BLK& B) {
    const int m = 2 * d, m2 = 4 * d;
    const int Kd = pad32(d), Km = pad32(m), Km2 = pad32(m2);
    k_ln<<<(P + 255) / 256, 256, 0, stream>>>(Xs, B.lg, B.lb, T16, P, d, Kd);
    CW c1 = mk(B.ff.c1.w, m, d, 1);
    conv(T16, c1, Y, P, H, W);
    bnact(Y, B.ff.c1.bn, P, m, Km, nullptr, T16, 0);
    CW c2 = mk(B.ff.c2.w, m2, m, 1);
    conv(T16, c2, Y, P, H, W);
    bnact(Y, B.ff.c2.bn, P, m2, Km2, nullptr, T16, 1 | 2);   // gelu + bact sign
    f16* w3 = mkg(B.ff.c3.w, m2);
    { long long n = (long long)P * m2;
      k_gconv3<<<(unsigned)((n + 255) / 256), 256, 0, stream>>>(T16, w3, Y, P, H, W, m2, Km2); }
    bnact(Y, B.ff.c3.bn, P, m2, Km2, nullptr, T16, 1);       // gelu
    CW c4 = mk(B.ff.c4.w, m, m2, 1);
    conv(T16, c4, Y, P, H, W);
    bnact(Y, B.ff.c4.bn, P, m, Km, nullptr, T16, 0);
    CW c5 = mk(B.ff.c5.w, d, m, 1);
    conv(T16, c5, Y, P, H, W);
    bnact(Y, B.ff.c5.bn, P, d, Kd, Y, nullptr, 0);           // f32 in place
    { long long n = (long long)P * d;
      k_add<<<(unsigned)((n + 255) / 256), 256, 0, stream>>>(Xs, Y, n); }
  };

  // --- embedding: fea = bconv3x3(sign(x)) ---
  { long long n = (long long)P0 * 32;
    k_sign_nchw<<<(unsigned)((n + 255) / 256), 256, 0, stream>>>(x_in, T16, P0, 28, 32, H0 * W0); }
  CW cemb = mk(emb_w, 28, 28, 3);
  conv(T16, cemb, X0, P0, H0, W0);

  // --- encoder stage 0 ---
  for (int i = 0; i < 2; ++i) run_blk(X0, P0, H0, W0, 28, e0b[i]);
  { long long n = (long long)P1 * 32;
    k_avgpool<<<(unsigned)((n + 255) / 256), 256, 0, stream>>>(X0, T16, P1, 28, 32, H1, W1); }
  CW cd0 = mk(down0.w, 56, 28, 3);
  conv(T16, cd0, Y, P1, H1, W1);
  bnact(Y, down0.bn, P1, 56, 56, X1, nullptr, 0);

  // --- encoder stage 1 ---
  for (int i = 0; i < 4; ++i) run_blk(X1, P1, H1, W1, 56, e1b[i]);
  { long long n = (long long)P2 * 64;
    k_avgpool<<<(unsigned)((n + 255) / 256), 256, 0, stream>>>(X1, T16, P2, 56, 64, H2, W2); }
  CW cd1 = mk(down1.w, 112, 56, 3);
  conv(T16, cd1, Y, P2, H2, W2);
  bnact(Y, down1.bn, P2, 112, 112, X2, nullptr, 0);

  // --- bottleneck ---
  for (int i = 0; i < 4; ++i) run_blk(X2, P2, H2, W2, 112, bo[i]);

  // --- decoder stage 0 ---
  { long long n = (long long)P1 * 128;
    k_up2<<<(unsigned)((n + 255) / 256), 256, 0, stream>>>(X2, T16, P1, 112, 128, H1, W1); }
  CW cu0 = mk(up0.w, 56, 112, 3);
  conv(T16, cu0, Y, P1, H1, W1);
  bnact(Y, up0.bn, P1, 56, 56, U, nullptr, 0);
  { long long n = (long long)P1 * 128;
    k_concat<<<(unsigned)((n + 255) / 256), 256, 0, stream>>>(U, X1, T16, P1, 56, 56, 128); }
  CW cf0 = mk(fus0.w, 56, 112, 1);
  conv(T16, cf0, Y, P1, H1, W1);
  bnact(Y, fus0.bn, P1, 56, 56, X1, nullptr, 0);
  for (int i = 0; i < 4; ++i) run_blk(X1, P1, H1, W1, 56, d0b[i]);

  // --- decoder stage 1 ---
  { long long n = (long long)P0 * 64;
    k_up2<<<(unsigned)((n + 255) / 256), 256, 0, stream>>>(X1, T16, P0, 56, 64, H0, W0); }
  CW cu1 = mk(up1.w, 28, 56, 3);
  conv(T16, cu1, Y, P0, H0, W0);
  bnact(Y, up1.bn, P0, 28, 28, U, nullptr, 0);
  { long long n = (long long)P0 * 64;
    k_concat<<<(unsigned)((n + 255) / 256), 256, 0, stream>>>(U, X0, T16, P0, 28, 28, 64); }
  CW cf1 = mk(fus1.w, 28, 56, 1);
  conv(T16, cf1, Y, P0, H0, W0);
  bnact(Y, fus1.bn, P0, 28, 28, X0, nullptr, 0);
  for (int i = 0; i < 2; ++i) run_blk(X0, P0, H0, W0, 28, d1b[i]);

  // --- mapping + residual ---
  { long long n = (long long)P0 * 32;
    k_sign_f32<<<(unsigned)((n + 255) / 256), 256, 0, stream>>>(X0, T16, P0, 28, 32); }
  CW cm = mk(map_w, 28, 28, 3);
  conv(T16, cm, Y, P0, H0, W0);
  { long long n = (long long)P0 * 28;
    k_final<<<(unsigned)((n + 255) / 256), 256, 0, stream>>>(Y, x_in, (float*)d_out, P0, 28, H0 * W0); }
}
